// AdditiveAttention_21766894256602
// MI455X (gfx1250) — matrix operand order repaired and checked by reading, not compiled, not run on hardware
//
#include <hip/hip_runtime.h>

typedef __attribute__((ext_vector_type(16))) _Float16 v16h;
typedef __attribute__((ext_vector_type(8)))  _Float16 v8h;
typedef __attribute__((ext_vector_type(4)))  _Float16 v4h;
typedef __attribute__((ext_vector_type(8)))  float    v8f;

#define B_    32
#define S_    2048
#define HLEN_ 1024
#define VLEN_ 1024
#define HID_  1024

// workspace layout (bytes)
#define WS_UPACK 0u                              // 64*32*32*16 halves = 2 MB
#define WS_HPROJ (2u << 20)                      // 32*1024 f32 = 128 KB
#define WS_E     (WS_HPROJ + 32u * 1024u * 4u)   // 32*2048 f32 = 256 KB
#define WS_STATS (WS_E + 32u * 2048u * 4u)       // 64 f32 (padded 1 KB)
#define WS_PART  (WS_STATS + 1024u)              // 32*16*1024 f32 = 2 MB

// branch-free tanh: tanh(x) = 1 - 2/(exp(2x)+1); exact limits at +-inf.
__device__ __forceinline__ float fast_tanh(float x) {
    float ex = __expf(2.0f * x);                            // v_exp_f32 path
    return 1.0f - 2.0f * __builtin_amdgcn_rcpf(ex + 1.0f);  // v_rcp_f32
}

// ---------------- kernel 1: h_proj = h @ W_w^T + W_b ----------------
__global__ void hproj_kernel(const float* __restrict__ h, const float* __restrict__ Ww,
                             const float* __restrict__ Wb, float* __restrict__ hproj) {
    int b = blockIdx.x;
    int d = blockIdx.y * blockDim.x + threadIdx.x;
    const float* hr = h + b * HLEN_;
    const float* wr = Ww + (size_t)d * HLEN_;
    float acc = Wb[d];
    for (int k = 0; k < HLEN_; k += 4) {
        float4 hk = *(const float4*)(hr + k);
        float4 wk = *(const float4*)(wr + k);
        acc += hk.x * wk.x + hk.y * wk.y + hk.z * wk.z + hk.w * wk.w;
    }
    hproj[b * HID_ + d] = acc;
}

// ---------------- kernel 2: pack U_w into f16 WMMA B-fragment order ----------------
// Upack[(((dt*32 + kt)*32 + lane)*16 + i)] = (f16) U_w[d, k]
//   d = dt*16 + (lane&15);  k = kt*32 + (lane>>4)*16 + i
__global__ void pack_kernel(const float* __restrict__ Uw, _Float16* __restrict__ up) {
    int id   = blockIdx.x * blockDim.x + threadIdx.x;   // 0 .. 1048575
    int i    = id & 15;
    int lane = (id >> 4) & 31;
    int kt   = (id >> 9) & 31;
    int dt   = id >> 14;
    int d = dt * 16 + (lane & 15);
    int k = kt * 32 + ((lane >> 4) << 4) + i;
    up[id] = (_Float16)Uw[(size_t)d * VLEN_ + k];
}

// ---------------- kernel 3: e[b,s] = w_w . tanh(hproj + V U^T) via WMMA ----------------
// 4 waves; each wave processes all 32 staged rows (two C tiles sharing one B
// fragment) over its own quarter of the d-tiles (B traffic halved vs row-split).
__global__ __launch_bounds__(128) void attn_e_kernel(
    const float* __restrict__ V, const _Float16* __restrict__ up,
    const float* __restrict__ hproj, const float* __restrict__ ww,
    float* __restrict__ e) {
    __shared__ _Float16 Vs[32 * VLEN_];   // 64 KB: 32 s-rows, f16, row-major
    __shared__ float ebuf[4][32];         // [d-quarter][local row]
    int b  = blockIdx.x >> 6;             // S_/32 = 64 chunks per batch
    int s0 = (blockIdx.x & 63) * 32;
    int tid = threadIdx.x;

    // stage 32x1024 f32 -> f16 LDS (float4 loads, 8-byte ds stores)
    const float* Vg = V + ((size_t)(b * S_ + s0)) * VLEN_;
    for (int f = tid; f < 32 * VLEN_ / 4; f += 128) {
        float4 x = *(const float4*)(Vg + (size_t)f * 4);
        v4h hv = {(_Float16)x.x, (_Float16)x.y, (_Float16)x.z, (_Float16)x.w};
        *(v4h*)(&Vs[f * 4]) = hv;
    }
    __syncthreads();

    int lane = tid & 31;
    int dq   = tid >> 5;          // d quarter: tiles dq*16 .. dq*16+15
    int half = lane >> 4;
    int col  = lane & 15;         // = A-row M within group, = C-column N

    const _Float16* arow0 = &Vs[col * VLEN_ + half * 16];         // rows 0-15
    const _Float16* arow1 = arow0 + 16 * VLEN_;                   // rows 16-31
    const v16h* bp = (const v16h*)up;

    float eacc0[8] = {0.f, 0.f, 0.f, 0.f, 0.f, 0.f, 0.f, 0.f};
    float eacc1[8] = {0.f, 0.f, 0.f, 0.f, 0.f, 0.f, 0.f, 0.f};

    for (int dt = dq * 16; dt < dq * 16 + 16; ++dt) {
        int d = dt * 16 + col;
        float hp  = hproj[b * HID_ + d];
        float wwd = ww[d];
        v8f c0 = {}, c1 = {};
#pragma unroll 2
        for (int kt = 0; kt < 32; ++kt) {
            // shared B fragment for both row groups
            v16h bb = bp[(dt * 32 + kt) * 32 + lane];
            // A fragments: lanes 0-15 K{0..15}, lanes 16-31 K{16..31}
            v8h a0lo = *(const v8h*)(arow0 + kt * 32);
            v8h a0hi = *(const v8h*)(arow0 + kt * 32 + 8);
            v16h a0 = __builtin_shufflevector(a0lo, a0hi, 0, 1, 2, 3, 4, 5, 6, 7,
                                              8, 9, 10, 11, 12, 13, 14, 15);
            v8h a1lo = *(const v8h*)(arow1 + kt * 32);
            v8h a1hi = *(const v8h*)(arow1 + kt * 32 + 8);
            v16h a1 = __builtin_shufflevector(a1lo, a1hi, 0, 1, 2, 3, 4, 5, 6, 7,
                                              8, 9, 10, 11, 12, 13, 14, 15);
            c0 = __builtin_amdgcn_wmma_f32_16x16x32_f16(
                     false, a0, false, bb, (short)0, c0, false, false);
            c1 = __builtin_amdgcn_wmma_f32_16x16x32_f16(
                     false, a1, false, bb, (short)0, c1, false, false);
        }
        // epilogue: per-lane d-column contribution, rows m = r + 8*half
#pragma unroll
        for (int r = 0; r < 8; ++r) {
            eacc0[r] += wwd * fast_tanh(c0[r] + hp);
            eacc1[r] += wwd * fast_tanh(c1[r] + hp);
        }
    }

    // reduce over the 16 d-columns (lanes within each half)
#pragma unroll
    for (int r = 0; r < 8; ++r) {
        float v0 = eacc0[r], v1 = eacc1[r];
        v0 += __shfl_xor(v0, 1, 32);  v1 += __shfl_xor(v1, 1, 32);
        v0 += __shfl_xor(v0, 2, 32);  v1 += __shfl_xor(v1, 2, 32);
        v0 += __shfl_xor(v0, 4, 32);  v1 += __shfl_xor(v1, 4, 32);
        v0 += __shfl_xor(v0, 8, 32);  v1 += __shfl_xor(v1, 8, 32);
        eacc0[r] = v0; eacc1[r] = v1;
    }
    if (col == 0) {   // lane 0 -> rows +0..7, lane 16 -> rows +8..15
#pragma unroll
        for (int r = 0; r < 8; ++r) {
            ebuf[dq][half * 8 + r]      = eacc0[r];   // rows 0-15
            ebuf[dq][16 + half * 8 + r] = eacc1[r];   // rows 16-31
        }
    }
    __syncthreads();
    if (tid < 32)
        e[b * S_ + s0 + tid] =
            (ebuf[0][tid] + ebuf[1][tid]) + (ebuf[2][tid] + ebuf[3][tid]);
}

// ---------------- kernel 4: softmax stats per batch row ----------------
__global__ void stats_kernel(const float* __restrict__ e, float* __restrict__ stats) {
    __shared__ float red[256];
    int b = blockIdx.x, tid = threadIdx.x;
    const float* er = e + b * S_;
    float mx = -1e30f;
    for (int s = tid; s < S_; s += 256) mx = fmaxf(mx, er[s]);
    red[tid] = mx; __syncthreads();
    for (int off = 128; off > 0; off >>= 1) {
        if (tid < off) red[tid] = fmaxf(red[tid], red[tid + off]);
        __syncthreads();
    }
    mx = red[0]; __syncthreads();
    float sum = 0.f;
    for (int s = tid; s < S_; s += 256) sum += expf(er[s] - mx);
    red[tid] = sum; __syncthreads();
    for (int off = 128; off > 0; off >>= 1) {
        if (tid < off) red[tid] += red[tid + off];
        __syncthreads();
    }
    if (tid == 0) { stats[b * 2] = mx; stats[b * 2 + 1] = 1.0f / red[0]; }
}

// ---------------- kernel 5: partial beta-weighted sums of V ----------------
__global__ void wsum_kernel(const float* __restrict__ V, const float* __restrict__ e,
                            const float* __restrict__ stats, float* __restrict__ part) {
    __shared__ float sbeta[128];
    int b = blockIdx.x, chunk = blockIdx.y, tid = threadIdx.x;
    int s0 = chunk * 128;
    float mx = stats[b * 2], inv = stats[b * 2 + 1];
    if (tid < 128) sbeta[tid] = expf(e[b * S_ + s0 + tid] - mx) * inv;
    __syncthreads();
    float a0 = 0.f, a1 = 0.f, a2 = 0.f, a3 = 0.f;
    const float* Vp = V + ((size_t)(b * S_ + s0)) * VLEN_;
    for (int ls = 0; ls < 128; ++ls) {
        float bsc = sbeta[ls];
        const float* r = Vp + (size_t)ls * VLEN_;
        a0 += bsc * r[tid];
        a1 += bsc * r[tid + 256];
        a2 += bsc * r[tid + 512];
        a3 += bsc * r[tid + 768];
    }
    float* pr = part + ((size_t)(b * 16 + chunk)) * VLEN_;
    pr[tid] = a0; pr[tid + 256] = a1; pr[tid + 512] = a2; pr[tid + 768] = a3;
}

// ---------------- kernel 6: fixed-order reduction of partials -> out ----------------
__global__ void reduce_kernel(const float* __restrict__ part, float* __restrict__ out) {
    int id = blockIdx.x * 256 + threadIdx.x;  // 0 .. 32767
    int b = id >> 10, v = id & 1023;
    float s = 0.f;
#pragma unroll
    for (int c = 0; c < 16; ++c)
        s += part[((size_t)(b * 16 + c)) * VLEN_ + v];
    out[id] = s;
}

extern "C" void kernel_launch(void* const* d_in, const int* in_sizes, int n_in,
                              void* d_out, int out_size, void* d_ws, size_t ws_size,
                              hipStream_t stream) {
    const float* h  = (const float*)d_in[0];
    const float* V  = (const float*)d_in[1];
    const float* Ww = (const float*)d_in[2];
    const float* Wb = (const float*)d_in[3];
    const float* Uw = (const float*)d_in[4];
    const float* ww = (const float*)d_in[5];
    char* ws = (char*)d_ws;
    _Float16* up = (_Float16*)(ws + WS_UPACK);
    float* hproj = (float*)(ws + WS_HPROJ);
    float* e     = (float*)(ws + WS_E);
    float* stats = (float*)(ws + WS_STATS);
    float* part  = (float*)(ws + WS_PART);
    float* out   = (float*)d_out;

    hipLaunchKernelGGL(hproj_kernel,  dim3(32, 8),  dim3(128), 0, stream, h, Ww, Wb, hproj);
    hipLaunchKernelGGL(pack_kernel,   dim3(4096),   dim3(256), 0, stream, Uw, up);
    hipLaunchKernelGGL(attn_e_kernel, dim3(2048),   dim3(128), 0, stream, V, up, hproj, ww, e);
    hipLaunchKernelGGL(stats_kernel,  dim3(32),     dim3(256), 0, stream, e, stats);
    hipLaunchKernelGGL(wsum_kernel,   dim3(32, 16), dim3(256), 0, stream, V, e, stats, part);
    hipLaunchKernelGGL(reduce_kernel, dim3(128),    dim3(256), 0, stream, part, out);
}
